// GLALayer_62302795596535
// MI455X (gfx1250) — matrix operand order repaired and checked by reading, not compiled, not run on hardware
//
#include <hip/hip_runtime.h>
#include <hip/hip_bf16.h>
#include <math.h>

typedef __bf16 bf16_t;
typedef __attribute__((ext_vector_type(16))) __bf16 v16bf;
typedef __attribute__((ext_vector_type(8)))  float  v8f;

// Problem constants
#define BB 2
#define LL 4096
#define DM 1024
#define HH 16
#define DK 64
#define DV 64
#define CC 128
#define SC 16
#define NCH (LL / CC)      // 32 chunks
#define NS (CC / SC)       // 8 sub-chunks
#define MROWS (BB * LL)    // 8192

#define LDP 40             // padded LDS row stride (elems) -> conflict-free

// ---------------------------------------------------------------------------
// WMMA helpers (CDNA5 gfx1250, wave32). Layouts per CDNA5 ISA.
// ---------------------------------------------------------------------------

__device__ __forceinline__ v8f wmma_bf16(v16bf a, v16bf b, v8f c) {
  return __builtin_amdgcn_wmma_f32_16x16x32_bf16(
      /*neg_a=*/false, a, /*neg_b=*/false, b,
      /*c_mod=*/(short)0, c, /*reuse_a=*/false, /*reuse_b=*/false);
}

// A fragment (16x32) from row-major bf16 tile, leading dim ld.
__device__ __forceinline__ v16bf a_frag_bf(const bf16_t* tile, int ld) {
  int lane = threadIdx.x & 31;
  int m  = lane & 15;
  int hi = (lane >> 4) << 4;
  const bf16_t* r = tile + (size_t)m * ld;
  v16bf f;
#pragma unroll
  for (int j = 0; j < 8; ++j) f[j] = r[hi + j];
#pragma unroll
  for (int j = 0; j < 8; ++j) f[8 + j] = r[hi + 8 + j];
  return f;
}

// B fragment (32x16): Bt is B^T row-major, leading dim ld.
__device__ __forceinline__ v16bf b_frag_bf(const bf16_t* Bt, int ld) {
  int lane = threadIdx.x & 31;
  int n    = lane & 15;
  int koff = (lane >> 4) << 4;
  const bf16_t* r = Bt + (size_t)n * ld + koff;
  v16bf f;
#pragma unroll
  for (int j = 0; j < 16; ++j) f[j] = r[j];
  return f;
}

// A fragment from fp32 tile with per-lane (per-row) scale, convert to bf16.
__device__ __forceinline__ v16bf a_frag_f32s(const float* tile, int ld, float s) {
  int lane = threadIdx.x & 31;
  int m  = lane & 15;
  int hi = (lane >> 4) << 4;
  const float* r = tile + (size_t)m * ld;
  v16bf f;
#pragma unroll
  for (int j = 0; j < 8; ++j) f[j] = (bf16_t)(r[hi + j] * s);
#pragma unroll
  for (int j = 0; j < 8; ++j) f[8 + j] = (bf16_t)(r[hi + 8 + j] * s);
  return f;
}

// B fragment from fp32 B^T tile with per-lane scale.
__device__ __forceinline__ v16bf b_frag_f32s(const float* Bt, int ld, float s) {
  int lane = threadIdx.x & 31;
  int n    = lane & 15;
  int koff = (lane >> 4) << 4;
  const float* r = Bt + (size_t)n * ld + koff;
  v16bf f;
#pragma unroll
  for (int j = 0; j < 16; ++j) f[j] = (bf16_t)(r[j] * s);
  return f;
}

__device__ __forceinline__ void store_c_f32(float* C, int ldc, const v8f& acc) {
  int lane = threadIdx.x & 31;
  int cn = lane & 15;
  int mo = (lane >> 4) << 3;
#pragma unroll
  for (int r = 0; r < 8; ++r) C[(size_t)(mo + r) * ldc + cn] = acc[r];
}

// LDS byte offset of a __shared__ object: generic LDS pointer low 32 bits
// are the in-group offset (ISA 10.2).
__device__ __forceinline__ unsigned lds_off_of(const void* p) {
  return (unsigned)(uintptr_t)p;
}

// Issue one per-lane 16B async global->LDS copy (GVS mode), ASYNCcnt-tracked.
__device__ __forceinline__ void async_g2l_b128(unsigned lds_off,
                                               unsigned gbyte_off,
                                               unsigned long long sbase) {
  asm volatile("global_load_async_to_lds_b128 %0, %1, %2"
               :: "v"(lds_off), "v"(gbyte_off), "s"(sbase)
               : "memory");
}

__device__ __forceinline__ void wait_async0() {
  asm volatile("s_wait_asynccnt 0" ::: "memory");
}

// ---------------------------------------------------------------------------
// Kernel: fp32 -> bf16 elementwise convert
// ---------------------------------------------------------------------------
__global__ void cvt_bf16_kernel(const float* __restrict__ in,
                                bf16_t* __restrict__ out,
                                unsigned long long n) {
  unsigned long long stride = (unsigned long long)gridDim.x * blockDim.x;
  for (unsigned long long i = (unsigned long long)blockIdx.x * blockDim.x + threadIdx.x;
       i < n; i += stride)
    out[i] = (bf16_t)in[i];
}

// ---------------------------------------------------------------------------
// Kernel: C[M,N] = A[M,K] * Bt[N,K]^T  (bf16 in, fp32 out)
// block = 256 thr = 8 waves; block tile 128x128; wave tile 32x64; K-step 32.
// Double-buffered LDS staging via global_load_async_to_lds_b128 (ASYNCcnt).
// ---------------------------------------------------------------------------
__global__ __launch_bounds__(256)
void gemm_bf16_kernel(const bf16_t* __restrict__ A,
                      const bf16_t* __restrict__ Bt,
                      float* __restrict__ C, int M, int N, int K) {
  __shared__ bf16_t Asm[2][128][LDP];   // A tile: rows of M, 32 K-cols (padded)
  __shared__ bf16_t Bsm[2][128][LDP];   // B^T tile: rows of N, 32 K-cols

  int tid = threadIdx.x;
  int wv = tid >> 5;
  int wm = wv >> 1;          // 0..3
  int wn = wv & 1;           // 0..1
  int bm = blockIdx.x * 128;
  int bn = blockIdx.y * 128;

  unsigned long long sA = (unsigned long long)(uintptr_t)(A + (size_t)bm * K);
  unsigned long long sB = (unsigned long long)(uintptr_t)(Bt + (size_t)bn * K);

  // Per-thread staging chunks: 512 x 16B per tile -> 2 chunks/thread/matrix.
  int c0 = tid;              // chunk ids tid and tid+256
  int r0 = c0 >> 2,        q0 = (c0 & 3) << 3;
  int r1 = (c0 + 256) >> 2, q1 = ((c0 + 256) & 3) << 3;

  v8f acc[2][4];
#pragma unroll
  for (int i = 0; i < 2; ++i)
#pragma unroll
    for (int j = 0; j < 4; ++j) acc[i][j] = (v8f)0.f;

  // Prologue: stage K-slice 0 into buffer 0.
  {
    unsigned g0 = (unsigned)((r0 * K + q0) * 2);
    unsigned g1 = (unsigned)((r1 * K + q1) * 2);
    async_g2l_b128(lds_off_of(&Asm[0][r0][q0]), g0, sA);
    async_g2l_b128(lds_off_of(&Asm[0][r1][q1]), g1, sA);
    async_g2l_b128(lds_off_of(&Bsm[0][r0][q0]), g0, sB);
    async_g2l_b128(lds_off_of(&Bsm[0][r1][q1]), g1, sB);
  }

  for (int kk = 0; kk < K; kk += 32) {
    int cur = (kk >> 5) & 1;
    wait_async0();           // my async copies for `cur` have landed
    __syncthreads();         // everyone's copies have landed

    if (kk + 32 < K) {       // stage next K-slice while computing on `cur`
      int nxt = cur ^ 1;
      unsigned g0 = (unsigned)((r0 * K + kk + 32 + q0) * 2);
      unsigned g1 = (unsigned)((r1 * K + kk + 32 + q1) * 2);
      async_g2l_b128(lds_off_of(&Asm[nxt][r0][q0]), g0, sA);
      async_g2l_b128(lds_off_of(&Asm[nxt][r1][q1]), g1, sA);
      async_g2l_b128(lds_off_of(&Bsm[nxt][r0][q0]), g0, sB);
      async_g2l_b128(lds_off_of(&Bsm[nxt][r1][q1]), g1, sB);
    }

    v16bf a0 = a_frag_bf(&Asm[cur][wm * 32][0], LDP);
    v16bf a1 = a_frag_bf(&Asm[cur][wm * 32 + 16][0], LDP);
#pragma unroll
    for (int nj = 0; nj < 4; ++nj) {
      v16bf bf = b_frag_bf(&Bsm[cur][wn * 64 + nj * 16][0], LDP);
      acc[0][nj] = wmma_bf16(a0, bf, acc[0][nj]);
      acc[1][nj] = wmma_bf16(a1, bf, acc[1][nj]);
    }
    __syncthreads();         // all waves done reading `cur` before refill
  }

#pragma unroll
  for (int mi = 0; mi < 2; ++mi)
#pragma unroll
    for (int nj = 0; nj < 4; ++nj)
      store_c_f32(C + (size_t)(bm + wm * 32 + mi * 16) * N + bn + wn * 64 + nj * 16,
                  N, acc[mi][nj]);
}

// ---------------------------------------------------------------------------
// Kernel: L2-normalize Q and K in place along each head's 64-dim vector
// ---------------------------------------------------------------------------
__global__ void normqk_kernel(float* __restrict__ Qf, float* __restrict__ Kf) {
  unsigned long long total = 2ull * MROWS * HH;
  unsigned long long stride = (unsigned long long)gridDim.x * blockDim.x;
  for (unsigned long long i = (unsigned long long)blockIdx.x * blockDim.x + threadIdx.x;
       i < total; i += stride) {
    float* base = ((i & 1ull) ? Kf : Qf) + (i >> 1) * 64;
    float s = 0.f;
#pragma unroll
    for (int d = 0; d < 64; ++d) s += base[d] * base[d];
    float inv = 1.f / fmaxf(sqrtf(s), 1e-12f);
#pragma unroll
    for (int d = 0; d < 64; ++d) base[d] *= inv;
  }
}

// ---------------------------------------------------------------------------
// Kernel: gate[b][h][l] = log_sigmoid(x[b,l,:] . Wg[h,:] + bg[h])
// ---------------------------------------------------------------------------
__global__ __launch_bounds__(128)
void gate_kernel(const float* __restrict__ x, const float* __restrict__ Wg,
                 const float* __restrict__ bg, float* __restrict__ gate) {
  __shared__ float red[128];
  size_t row = blockIdx.x;               // b*L + l
  int h = threadIdx.x >> 3;
  int sub = threadIdx.x & 7;
  const float* xr = x + row * DM;
  const float* wr = Wg + (size_t)h * DM;
  float s = 0.f;
  for (int k = sub; k < DM; k += 8) s += xr[k] * wr[k];
  red[threadIdx.x] = s;
  __syncthreads();
  if (sub == 0) {
    float z = bg[h];
#pragma unroll
    for (int j = 0; j < 8; ++j) z += red[(h << 3) + j];
    float ls = fminf(z, 0.f) - log1pf(__expf(-fabsf(z)));   // stable log_sigmoid
    size_t b = row / LL, l = row % LL;
    gate[((b * HH + h) * (size_t)LL) + l] = ls;
  }
}

// ---------------------------------------------------------------------------
// Kernel: per-chunk inclusive cumsum of gate -> g (Hillis-Steele over 128)
// ---------------------------------------------------------------------------
__global__ __launch_bounds__(128)
void cumsum_kernel(const float* __restrict__ gate, float* __restrict__ g) {
  __shared__ float sb[128];
  int idx = blockIdx.x;
  int bh = idx / NCH;
  int n  = idx % NCH;
  int t = threadIdx.x;
  size_t base = (size_t)bh * LL + (size_t)n * CC;
  sb[t] = gate[base + t];
  __syncthreads();
#pragma unroll
  for (int off = 1; off < 128; off <<= 1) {
    float v = (t >= off) ? sb[t - off] : 0.f;
    __syncthreads();
    sb[t] += v;
    __syncthreads();
  }
  g[base + t] = sb[t];
}

// ---------------------------------------------------------------------------
// Attention kernel: one block per (b,h,chunk). 256 threads = 8 waves;
// wave i owns sub-chunk row i (16 tokens). All matmuls via bf16 WMMA.
// ---------------------------------------------------------------------------
__global__ __launch_bounds__(256)
void attn_kernel(const float* __restrict__ Qf, const float* __restrict__ Kf,
                 const float* __restrict__ Vf, const float* __restrict__ g,
                 bf16_t* __restrict__ Obf) {
  __shared__ float  gsh[CC];
  __shared__ bf16_t vT[DV][CC];          // V transposed: vT[dcol][token]
  __shared__ bf16_t stile[8][16 * 32];   // per-wave 16x32 bf16 score scratch

  int idx = blockIdx.x;
  int n = idx & (NCH - 1);
  int h = (idx >> 5) & (HH - 1);
  int b = idx >> 9;
  int tid = threadIdx.x;

  size_t cbase = ((size_t)(b * LL + n * CC)) * DM + (size_t)h * DK;
  size_t gbase = ((size_t)(b * HH + h)) * LL + (size_t)n * CC;

  if (tid < CC) gsh[tid] = g[gbase + tid];
  for (int ii = tid; ii < CC * DV; ii += 256) {
    int t = ii >> 6, d = ii & 63;
    vT[d][t] = (bf16_t)Vf[cbase + (size_t)t * DM + d];
  }
  __syncthreads();

  int wv = tid >> 5;
  int lane = tid & 31;
  int i = wv;                                 // sub-chunk row owned by wave
  float gl_i = gsh[i * SC + SC - 1];
  float gf_i = gsh[i * SC];

  // Qs fragments: per-lane row t -> scale exp(g_t - gl_i). Invariant over j.
  int am = lane & 15;
  float a_scale = __expf(gsh[i * SC + am] - gl_i);
  const float* Atile = Qf + cbase + (size_t)(i * SC) * DM;
  v16bf qa0 = a_frag_f32s(Atile + 0, DM, a_scale);
  v16bf qa1 = a_frag_f32s(Atile + 32, DM, a_scale);

  v8f oacc[4];
#pragma unroll
  for (int t = 0; t < 4; ++t) oacc[t] = (v8f)0.f;

  int sn = lane & 15;
  int mo = (lane >> 4) << 3;
  bf16_t* myt = &stile[wv][0];                // 16 x 32, ld = 32

  for (int jj = 0; jj < NS; jj += 2) {
#pragma unroll
    for (int jh = 0; jh < 2; ++jh) {
      int j = jj + jh;
      if (j > i) {                            // future sub-chunk: zeros
#pragma unroll
        for (int r = 0; r < 8; ++r)
          myt[(mo + r) * 32 + jh * 16 + sn] = (bf16_t)0.f;
      } else {
        float gl_j = gsh[j * SC + SC - 1];
        float gf_j = gsh[j * SC];
        float b_scale = __expf(gsh[j * SC + sn] - gf_j);
        const float* Btile = Kf + cbase + (size_t)(j * SC) * DM;
        v16bf kb0 = b_frag_f32s(Btile + 0, DM, b_scale);
        v16bf kb1 = b_frag_f32s(Btile + 32, DM, b_scale);
        v8f sacc = (v8f)0.f;
        sacc = wmma_bf16(qa0, kb0, sacc);
        sacc = wmma_bf16(qa1, kb1, sacc);
        if (j < i) {                          // inter: exp(gl_i - gl_j)
          float f = __expf(gl_i - gl_j);
#pragma unroll
          for (int r = 0; r < 8; ++r)
            myt[(mo + r) * 32 + jh * 16 + sn] = (bf16_t)(sacc[r] * f);
        } else {                              // diag: Dmask = exp(g_t-g_s)*tril
          float corr = __expf(gl_i + gf_i - 2.f * gsh[j * SC + sn]);
#pragma unroll
          for (int r = 0; r < 8; ++r) {
            int ti = mo + r;
            float v = (ti >= sn) ? sacc[r] * corr : 0.f;
            myt[(mo + r) * 32 + jh * 16 + sn] = (bf16_t)v;
          }
        }
      }
    }
    // order the per-wave LDS stores before fragment reload
    asm volatile("s_wait_dscnt 0" ::: "memory");

    v16bf sa = a_frag_bf(myt, 32);            // S row-tile i, K = tokens jj*16..+31
#pragma unroll
    for (int nt = 0; nt < 4; ++nt) {
      v16bf vb = b_frag_bf(&vT[nt * 16][jj * 16], CC);
      oacc[nt] = wmma_bf16(sa, vb, oacc[nt]);
    }
  }

  // store O (bf16) to [b*L + l][H*DV]
#pragma unroll
  for (int nt = 0; nt < 4; ++nt)
#pragma unroll
    for (int r = 0; r < 8; ++r) {
      int tok = i * SC + mo + r;
      Obf[cbase + (size_t)tok * DM + nt * 16 + sn] = (bf16_t)oacc[nt][r];
    }
}

// ---------------------------------------------------------------------------
// Host-side launcher
// ---------------------------------------------------------------------------
extern "C" void kernel_launch(void* const* d_in, const int* in_sizes, int n_in,
                              void* d_out, int out_size, void* d_ws, size_t ws_size,
                              hipStream_t stream) {
  (void)in_sizes; (void)n_in; (void)out_size; (void)ws_size;
  const float* x  = (const float*)d_in[0];
  const float* Wq = (const float*)d_in[1];
  const float* Wk = (const float*)d_in[2];
  const float* Wv = (const float*)d_in[3];
  const float* Wg = (const float*)d_in[4];
  const float* bg = (const float*)d_in[5];
  const float* Wo = (const float*)d_in[6];
  float* out = (float*)d_out;

  char* ws = (char*)d_ws;
  constexpr size_t SZ_XBF = (size_t)MROWS * DM * 2;
  constexpr size_t SZ_WBF = (size_t)DM * DM * 2;
  constexpr size_t SZ_F   = (size_t)MROWS * DM * 4;
  constexpr size_t SZ_G   = (size_t)BB * HH * LL * 4;

  bf16_t* xbf  = (bf16_t*)(ws);                         size_t off = SZ_XBF;
  bf16_t* wqbf = (bf16_t*)(ws + off); off += SZ_WBF;
  bf16_t* wkbf = (bf16_t*)(ws + off); off += SZ_WBF;
  bf16_t* wvbf = (bf16_t*)(ws + off); off += SZ_WBF;
  bf16_t* wobf = (bf16_t*)(ws + off); off += SZ_WBF;
  float*  Qf   = (float*)(ws + off);  off += SZ_F;
  float*  Kf   = (float*)(ws + off);  off += SZ_F;
  float*  Vf   = (float*)(ws + off);  off += SZ_F;
  float*  gate = (float*)(ws + off);  off += SZ_G;
  float*  gcs  = (float*)(ws + off);  off += SZ_G;
  bf16_t* Obf  = (bf16_t*)(ws + off); off += SZ_XBF;

  // 1) convert inputs to bf16
  cvt_bf16_kernel<<<1024, 256, 0, stream>>>(x,  xbf,  (unsigned long long)MROWS * DM);
  cvt_bf16_kernel<<<256, 256, 0, stream>>>(Wq, wqbf, (unsigned long long)DM * DM);
  cvt_bf16_kernel<<<256, 256, 0, stream>>>(Wk, wkbf, (unsigned long long)DM * DM);
  cvt_bf16_kernel<<<256, 256, 0, stream>>>(Wv, wvbf, (unsigned long long)DM * DM);
  cvt_bf16_kernel<<<256, 256, 0, stream>>>(Wo, wobf, (unsigned long long)DM * DM);

  // 2) QKV projections (WMMA + async LDS pipeline)
  dim3 ggrid(MROWS / 128, DM / 128);
  gemm_bf16_kernel<<<ggrid, 256, 0, stream>>>(xbf, wqbf, Qf, MROWS, DM, DM);
  gemm_bf16_kernel<<<ggrid, 256, 0, stream>>>(xbf, wkbf, Kf, MROWS, DM, DM);
  gemm_bf16_kernel<<<ggrid, 256, 0, stream>>>(xbf, wvbf, Vf, MROWS, DM, DM);

  // 3) L2 norm of Q,K per head vector
  normqk_kernel<<<1024, 256, 0, stream>>>(Qf, Kf);

  // 4) gates + per-chunk cumsum
  gate_kernel<<<MROWS, 128, 0, stream>>>(x, Wg, bg, gate);
  cumsum_kernel<<<BB * HH * NCH, 128, 0, stream>>>(gate, gcs);

  // 5) chunk-local gated attention (WMMA), writes bf16 O
  attn_kernel<<<BB * HH * NCH, 256, 0, stream>>>(Qf, Kf, Vf, gcs, Obf);

  // 6) output projection (WMMA) -> fp32 d_out
  gemm_bf16_kernel<<<ggrid, 256, 0, stream>>>(Obf, wobf, out, MROWS, DM, DM);
}
